// GCLSTM_4320737100309
// MI455X (gfx1250) — matrix operand order repaired and checked by reading, not compiled, not run on hardware
//
#include <hip/hip_runtime.h>
#include <hip/hip_bf16.h>

typedef __attribute__((ext_vector_type(16))) __bf16 v16bf;
typedef __attribute__((ext_vector_type(8)))  float  v8f;
#define LDS_AS __attribute__((address_space(3)))

__device__ __forceinline__ unsigned short f2bf(float f) {
    unsigned int u = __float_as_uint(f);
    u += 0x7FFFu + ((u >> 16) & 1u);          // round-to-nearest-even
    return (unsigned short)(u >> 16);
}

// ---------------- Weight folding: WbigT[4][128][512] bf16 (K-transposed), bias[512] ----
// k rows:  [0,128)   = (Wp^T Wx[g])[f,e]      (x_t part)
//          [128,512) = Theta[g][hop][d,e]     (T0=H, T1, T2 parts)
// stored transposed: wbigT[((g*128+e) * 512) + k]  -> B tiles are contiguous in k
__global__ void build_wbig_k(const float* __restrict__ Wp, const float* __restrict__ Wx,
                             const float* __restrict__ bx, const float* __restrict__ Theta,
                             const float* __restrict__ cb, unsigned short* __restrict__ wbigT,
                             float* __restrict__ bias) {
    int idx = blockIdx.x * blockDim.x + threadIdx.x;   // 0 .. 512*512-1
    int r = idx >> 9, c = idx & 511;                   // r = k row, c = gate*128+e
    int g = c >> 7, e = c & 127;
    float val;
    if (r < 128) {
        float s = 0.f;
        const float* wxg = Wx + (size_t)g * 128 * 128 + e;
        #pragma unroll 4
        for (int d = 0; d < 128; ++d)
            s += Wp[d * 128 + r] * wxg[(size_t)d * 128];
        val = s;
    } else {
        int hop = (r - 128) >> 7;
        int rr  = (r - 128) & 127;
        val = Theta[(((size_t)g * 3 + hop) * 128 + rr) * 128 + e];
    }
    wbigT[(size_t)c * 512 + r] = f2bf(val);
    if (r == 0) bias[c] = bx[g * 128 + e] + cb[g * 128 + e];
}

// ---------------- Graph normalization ----------------
__global__ void deg_k(const long long* __restrict__ ei, float* __restrict__ deg, int E) {
    int e = blockIdx.x * blockDim.x + threadIdx.x;
    if (e < E) atomicAdd(&deg[(int)ei[e]], 1.0f);
}

__global__ void dinv_k(const float* __restrict__ deg, float* __restrict__ dinv, int N) {
    int n = blockIdx.x * blockDim.x + threadIdx.x;
    if (n < N) {
        float d = deg[n];
        dinv[n] = (d > 0.f) ? rsqrtf(fmaxf(d, 1.f)) : 0.f;
    }
}

__global__ void ew_k(const long long* __restrict__ ei, const float* __restrict__ dinv,
                     float* __restrict__ w, int E) {
    int e = blockIdx.x * blockDim.x + threadIdx.x;
    if (e < E) w[e] = -dinv[(int)ei[e]] * dinv[(int)ei[E + e]];
}

__global__ void negcopy_k(const float* __restrict__ h, float* __restrict__ t2, int n) {
    int i = blockIdx.x * blockDim.x + threadIdx.x;
    if (i < n) t2[i] = -h[i];
}

// out[dst] += scale * w[e] * h[src] ; one edge handled by 128 threads
__global__ void prop_k(const long long* __restrict__ ei, const float* __restrict__ w,
                       const float* __restrict__ h, float* __restrict__ outv,
                       float scale, int E) {
    int tid = threadIdx.x;
    int e = blockIdx.x * 2 + (tid >> 7);
    int f = tid & 127;
    if (e < E) {
        int src = (int)ei[e];
        int dst = (int)ei[E + e];
        float val = scale * w[e] * h[(size_t)src * 128 + f];
        atomicAdd(&outv[(size_t)dst * 128 + f], val);
    }
}

// ---------------- WMMA GEMM: pre[g][N][128] = [x|H|T1|T2] @ Wbig + bias ----------------
#define AS 34   // A LDS row stride (ushorts): 17 dwords, coprime with 64 banks
#define BS 40   // B LDS row stride (ushorts): 80B rows -> 16B-aligned async B128 stores

__global__ __launch_bounds__(256)
void gemm_wmma_k(const float* __restrict__ Xt, const float* __restrict__ H,
                 const float* __restrict__ T1, const float* __restrict__ T2,
                 const unsigned short* __restrict__ wbigT, const float* __restrict__ bias,
                 float* __restrict__ pre, int N) {
    __shared__ unsigned short As [128 * AS];   // A tile  : [row][k]  128x32 bf16
    __shared__ unsigned short Bst[128 * BS];   // B tile^T: [col][k]  128x32 bf16

    const int tid   = threadIdx.x;
    const int lane  = tid & 31;
    const int wave  = tid >> 5;
    const int wr    = wave & 3;     // row group: 32 rows each
    const int wc    = wave >> 2;    // col group: 64 cols each
    const int rlo   = lane & 15;
    const int khalf = lane >> 4;
    const int rowblk = blockIdx.x * 128;
    const int gate   = blockIdx.y;
    const int cbase  = gate * 128;

    // 32-bit LDS byte address base for the async-copy destination
    LDS_AS unsigned short* bstp = (LDS_AS unsigned short*)(&Bst[0]);

    v8f acc[2][4];
    #pragma unroll
    for (int i = 0; i < 2; ++i)
        #pragma unroll
        for (int j = 0; j < 4; ++j)
            acc[i][j] = v8f{0.f,0.f,0.f,0.f,0.f,0.f,0.f,0.f};

    for (int k0 = 0; k0 < 512; k0 += 32) {
        const float* A = (k0 < 128) ? Xt : (k0 < 256) ? H : (k0 < 384) ? T1 : T2;
        const int kin = k0 & 127;

        __syncthreads();

        // stage B: async global->LDS (CDNA5 path, no VGPR roundtrip).
        // 128 rows x 64 bytes; each thread issues two 16B copies.
        {
            const unsigned short* gb = wbigT + (size_t)cbase * 512 + k0;
            #pragma unroll
            for (int it = 0; it < 2; ++it) {
                int lin = it * 256 + tid;        // 0..511
                int c   = lin >> 2;
                int seg = lin & 3;
                const void* gaddr = (const void*)(gb + (size_t)c * 512 + seg * 8);
                LDS_AS unsigned short* laddr = bstp + c * BS + seg * 8;
                asm volatile("global_load_async_to_lds_b128 %0, %1, off"
                             :: "v"(laddr), "v"(gaddr)
                             : "memory");
            }
        }

        // stage A: 128 rows x 32 k (f32 -> bf16 in VALU)
        #pragma unroll
        for (int it = 0; it < 4; ++it) {
            int lin = it * 256 + tid;        // 0..1023 float4s
            int row = lin >> 3;
            int c4  = (lin & 7) * 4;
            int grow = rowblk + row;
            float4 v = make_float4(0.f, 0.f, 0.f, 0.f);
            if (grow < N) v = *(const float4*)(A + (size_t)grow * 128 + kin + c4);
            unsigned short* p = &As[row * AS + c4];
            p[0] = f2bf(v.x); p[1] = f2bf(v.y); p[2] = f2bf(v.z); p[3] = f2bf(v.w);
        }

        asm volatile("s_wait_asynccnt 0x0" ::: "memory");
        __syncthreads();

        union U { unsigned int u[8]; v16bf v; };
        v16bf afr[2], bfr[4];
        // A fragments: lane row = rlo, VGPR p holds K = khalf*16 + {2p, 2p+1}
        #pragma unroll
        for (int i = 0; i < 2; ++i) {
            U ua;
            const unsigned short* base = &As[(wr * 32 + i * 16 + rlo) * AS];
            #pragma unroll
            for (int p = 0; p < 4; ++p) {
                ua.u[p]     = *(const unsigned int*)(base + khalf * 16 + 2 * p);
                ua.u[4 + p] = *(const unsigned int*)(base + khalf * 16 + 8 + 2 * p);
            }
            afr[i] = ua.v;
        }
        // B fragments: lane col = rlo, VGPR p holds K = khalf*16 + {2p, 2p+1}
        #pragma unroll
        for (int j = 0; j < 4; ++j) {
            U ub;
            const unsigned short* base = &Bst[(wc * 64 + j * 16 + rlo) * BS + khalf * 16];
            #pragma unroll
            for (int p = 0; p < 8; ++p)
                ub.u[p] = *(const unsigned int*)(base + 2 * p);
            bfr[j] = ub.v;
        }
        #pragma unroll
        for (int i = 0; i < 2; ++i)
            #pragma unroll
            for (int j = 0; j < 4; ++j)
                acc[i][j] = __builtin_amdgcn_wmma_f32_16x16x32_bf16(
                    false, afr[i], false, bfr[j], (short)0, acc[i][j], false, false);
    }

    // epilogue: C/D layout — VGPR v, lanes 0-15: M=v, lanes 16-31: M=v+8, N=lane%16
    #pragma unroll
    for (int i = 0; i < 2; ++i) {
        #pragma unroll
        for (int j = 0; j < 4; ++j) {
            int col = wc * 64 + j * 16 + rlo;
            float b = bias[cbase + col];
            #pragma unroll
            for (int v = 0; v < 8; ++v) {
                int grow = rowblk + wr * 32 + i * 16 + v + 8 * khalf;
                if (grow < N)
                    pre[((size_t)gate * N + grow) * 128 + col] = acc[i][j][v] + b;
            }
        }
    }
}

// ---------------- LSTM pointwise ----------------
__global__ void lstm_k(const float* __restrict__ pre, float* __restrict__ Cst,
                       float* __restrict__ Hst, float* __restrict__ out,
                       int N, int t, int T) {
    int i = blockIdx.x * blockDim.x + threadIdx.x;
    if (i >= N * 128) return;
    size_t ND = (size_t)N * 128;
    float pi = pre[i], pf = pre[ND + i], pg = pre[2 * ND + i], po = pre[3 * ND + i];
    float I  = 1.f / (1.f + __expf(-pi));
    float Fg = 1.f / (1.f + __expf(-pf));
    float G  = tanhf(pg);
    float O  = 1.f / (1.f + __expf(-po));
    float c  = Fg * Cst[i] + I * G;
    float h  = O * tanhf(c);
    Cst[i] = c;
    Hst[i] = h;
    int n = i >> 7, e = i & 127;
    out[((size_t)n * T + t) * 128 + e] = h;
}

extern "C" void kernel_launch(void* const* d_in, const int* in_sizes, int n_in,
                              void* d_out, int out_size, void* d_ws, size_t ws_size,
                              hipStream_t stream) {
    const int T = 6, N = 50000, E = 800000;
    const float*     x     = (const float*)d_in[0];
    const long long* ei    = (const long long*)d_in[1];
    const float*     Wp    = (const float*)d_in[2];
    const float*     Wx    = (const float*)d_in[3];
    const float*     bx    = (const float*)d_in[4];
    const float*     Theta = (const float*)d_in[5];
    const float*     cb    = (const float*)d_in[6];
    float*           out   = (float*)d_out;

    char* ws = (char*)d_ws;
    size_t off = 0;
    auto take = [&](size_t bytes) { char* p = ws + off; off += (bytes + 255) & ~(size_t)255; return p; };

    unsigned short* wbigT = (unsigned short*)take((size_t)512 * 512 * 2);
    float* bias = (float*)take(512 * 4);
    float* deg  = (float*)take((size_t)N * 4);
    float* dinv = (float*)take((size_t)N * 4);
    float* w    = (float*)take((size_t)E * 4);
    float* Hb   = (float*)take((size_t)N * 128 * 4);
    float* Cb   = (float*)take((size_t)N * 128 * 4);
    float* T1   = (float*)take((size_t)N * 128 * 4);
    float* T2   = (float*)take((size_t)N * 128 * 4);
    float* pre  = (float*)take((size_t)4 * N * 128 * 4);
    (void)ws_size; (void)in_sizes; (void)n_in; (void)out_size;

    build_wbig_k<<<(512 * 512) / 256, 256, 0, stream>>>(Wp, Wx, bx, Theta, cb, wbigT, bias);
    hipMemsetAsync(Hb, 0, (size_t)N * 128 * 4, stream);
    hipMemsetAsync(Cb, 0, (size_t)N * 128 * 4, stream);

    const int gE   = (E + 255) / 256;
    const int gN   = (N + 255) / 256;
    const int gND  = (N * 128 + 255) / 256;
    const int gEp  = (E + 1) / 2;
    dim3 gemm_grid((N + 127) / 128, 4);

    for (int t = 0; t < T; ++t) {
        const long long* eit = ei + (size_t)t * 2 * E;
        const float*     xt  = x  + (size_t)t * N * 128;

        hipMemsetAsync(deg, 0, (size_t)N * 4, stream);
        deg_k <<<gE, 256, 0, stream>>>(eit, deg, E);
        dinv_k<<<gN, 256, 0, stream>>>(deg, dinv, N);
        ew_k  <<<gE, 256, 0, stream>>>(eit, dinv, w, E);

        hipMemsetAsync(T1, 0, (size_t)N * 128 * 4, stream);
        prop_k<<<gEp, 256, 0, stream>>>(eit, w, Hb, T1, 1.0f, E);     // T1 = L_hat H
        negcopy_k<<<gND, 256, 0, stream>>>(Hb, T2, N * 128);          // T2 = -H
        prop_k<<<gEp, 256, 0, stream>>>(eit, w, T1, T2, 2.0f, E);     // T2 = 2 L_hat T1 - H

        gemm_wmma_k<<<gemm_grid, 256, 0, stream>>>(xt, Hb, T1, T2, wbigT, bias, pre, N);
        lstm_k<<<gND, 256, 0, stream>>>(pre, Cb, Hb, out, N, t, T);
    }
}
